// ArDCA_48430051230491
// MI455X (gfx1250) — matrix operand order repaired and checked by reading, not compiled, not run on hardware
//
#include <hip/hip_runtime.h>
#include <hip/hip_bf16.h>

typedef __attribute__((ext_vector_type(16))) _Float16 v16h;
typedef __attribute__((ext_vector_type(8)))  float    v8f;

#define M_TOT 8192
#define LL    256
#define QW    21
#define NK    5376                 // L*Q (multiple of both 16 and 32)
#define MC    1024                 // M rows per chunk
#define PREP_BLOCKS 7056           // 5376*5376 / (256 threads * 16 elems)

static_assert(NK == LL * QW, "dims");
static_assert((long long)NK * NK == (long long)PREP_BLOCKS * 256 * 16, "prep grid exact");

// ---------------------------------------------------------------------------
// K1: permute+mask+convert  J[i][j][a][b] (f32) -> Jh[(i*Q+a)][(j*Q+b)] (f16),
//     zero where j >= i; accumulate regJ = sum(masked J^2) per-block (f32).
// ---------------------------------------------------------------------------
__global__ __launch_bounds__(256) void ardca_prep(const float* __restrict__ J,
                                                  _Float16* __restrict__ Jh,
                                                  float* __restrict__ partJ) {
  __shared__ float red[256];
  const int tid = threadIdx.x;
  const long long base = ((long long)blockIdx.x * 256 + tid) * 16;
  const int r  = (int)(base / NK);          // row  = i*Q + a  (16 elems never cross a row)
  const int c0 = (int)(base - (long long)r * NK);
  const int i  = r / QW;
  const int a  = r - QW * i;
  float s = 0.f;
#pragma unroll
  for (int t = 0; t < 16; ++t) {
    const int c = c0 + t;                   // col = j*Q + b
    const int j = c / QW;
    const int b = c - QW * j;
    float v = 0.f;
    if (j < i) v = J[(long long)((i * LL + j) * QW + a) * QW + b];
    s += v * v;
    Jh[base + t] = (_Float16)v;
  }
  red[tid] = s;
  __syncthreads();
  for (int off = 128; off > 0; off >>= 1) {
    if (tid < off) red[tid] += red[tid + off];
    __syncthreads();
  }
  if (tid == 0) partJ[blockIdx.x] = red[0];
}

// ---------------------------------------------------------------------------
// K2: WMMA GEMM  pair[mm][n] = sum_k onehot(X)[m][k] * Jh[n][k]
//     One wave -> 16(M) x 128(N) = 8 accumulators; 8 waves stacked in M.
//     A fragment (one-hot) built divisionless: a 32-wide K chunk holds at most
//     3 hot positions p_w = X[m, j0+w] + 21*w - rem  (rem = k0 % 21, kept
//     incrementally as a scalar). Each lane sets bits of a 16-bit hit mask and
//     expands it to the v16h fragment. One A build feeds 8 WMMAs.
// ---------------------------------------------------------------------------
__global__ __launch_bounds__(256) void ardca_gemm(const int* __restrict__ X,
                                                  const _Float16* __restrict__ Jh,
                                                  float* __restrict__ pairc,
                                                  int m0) {
  const int lane  = threadIdx.x & 31;
  const int wave  = threadIdx.x >> 5;
  const unsigned lhalf = (unsigned)(lane >> 4);     // which 16-lane half
  const int l16   = lane & 15;
  const int n0w   = blockIdx.x * 128;               // N-tile base (8 x 16 cols)
  const int mloc  = blockIdx.y * 128 + wave * 16;   // chunk-local M base of this wave
  const int mrow  = m0 + mloc + l16;                // global m held by this lane (A row)
  const int* Xrow = X + mrow * LL;

  v8f acc[8] = {};

  // largest i in this tile needs j < i  ->  k < imax*21 (Jh zero past mask)
  const int imax = (n0w + 127) / QW;
  int kmax = (imax * QW + 31) & ~31;
  if (kmax > NK) kmax = NK;

  int j0 = 0;                                       // k0 / 21
  int rem = 0;                                      // k0 % 21
  for (int k0 = 0; k0 < kmax; k0 += 32) {
    // ---- hot positions of the one-hot row inside this chunk ----
    const int jc1 = (j0 + 1 < LL) ? j0 + 1 : LL - 1;    // clamp: phantom j rejected by p<32
    const int jc2 = (j0 + 2 < LL) ? j0 + 2 : LL - 1;
    const int x0 = Xrow[j0];
    const int x1 = Xrow[jc1];
    const int x2 = Xrow[jc2];
    const int p0 = x0 - rem;                 // (-21, 21)
    const int p1 = x1 + QW - rem;            // (  0, 42)
    const int p2 = x2 + 2 * QW - rem;        // ( 21, 63)

    // ---- 16-bit hit mask for this lane's K slots ----
    // A lane layout: lanes0-15 hold K = 0..15 ; lanes16-31 hold 16..31
    // offset p belongs to this lane iff (p>>4)&1 == lhalf; slot = p&15
    unsigned mask = 0;
    {
      const unsigned u0 = (unsigned)p0, u1 = (unsigned)p1, u2 = (unsigned)p2;
      mask |= (u0 < 32u && (((u0 >> 4) & 1u) == lhalf)) ? (1u << (u0 & 15u)) : 0u;
      mask |= (u1 < 32u && (((u1 >> 4) & 1u) == lhalf)) ? (1u << (u1 & 15u)) : 0u;
      mask |= (u2 < 32u && (((u2 >> 4) & 1u) == lhalf)) ? (1u << (u2 & 15u)) : 0u;
    }

    // ---- expand mask -> v16h one-hot fragment (1.0h = 0x3C00) ----
    union { v16h v; unsigned u[8]; } af;
#pragma unroll
    for (int q = 0; q < 8; ++q) {
      af.u[q] = ((mask >> (2 * q)) & 1u) * 0x00003C00u
              + ((mask >> (2 * q + 1)) & 1u) * 0x3C000000u;
    }

    // ---- 8 N-subtiles share the A fragment ----
    // B lane layout: lane%16 = n, lane-half selects K 0..15 / 16..31 ->
    // 16 consecutive halfs of the (N-major) Jh row = one aligned 32B load.
    const _Float16* bbase = Jh + (long long)(n0w + l16) * NK + (k0 + 16 * (int)lhalf);
    __builtin_prefetch(bbase + 32, 0, 1);    // next K-chunk (speculative, L2-resident)
#pragma unroll
    for (int tt = 0; tt < 8; ++tt) {
      v16h bfrag = *(const v16h*)(bbase + (long long)(16 * tt) * NK);
      acc[tt] = __builtin_amdgcn_wmma_f32_16x16x32_f16(
          false, af.v, false, bfrag, (short)0, acc[tt], false, false);
    }

    // ---- advance k0/21, k0%21 without division ----
    rem += 32 - QW;  j0 += 1;                // rem in [11,32)
    if (rem >= QW) { rem -= QW; j0 += 1; }   // back to [0,21)
  }

  // D layout: VGPR r, lanes0-15 -> M=r ; lanes16-31 -> M=8+r ; N = lane&15
  float* outb = pairc + (long long)(mloc + (lhalf ? 8 : 0)) * NK;
#pragma unroll
  for (int tt = 0; tt < 8; ++tt) {
    const int col = n0w + 16 * tt + l16;
#pragma unroll
    for (int r = 0; r < 8; ++r)
      outb[(long long)r * NK + col] = acc[tt][r];
  }
}

// ---------------------------------------------------------------------------
// K3: per-(m,i) log-softmax over 21 states, gold log-prob, -W[m]-weighted sum.
//     Fixed-order block reduction (deterministic, no float atomics).
// ---------------------------------------------------------------------------
__global__ __launch_bounds__(256) void ardca_epi(const float* __restrict__ pairc,
                                                 const float* __restrict__ h,
                                                 const int* __restrict__ X,
                                                 const float* __restrict__ W,
                                                 float* __restrict__ nllPart,
                                                 int m0) {
  __shared__ float red[256];
  const int tid = threadIdx.x;
  const int idx = blockIdx.x * 256 + tid;
  const int mm  = idx >> 8;                  // / LL (LL == 256)
  const int i   = idx & (LL - 1);
  const int m   = m0 + mm;
  const float* pr = pairc + (long long)mm * NK + i * QW;
  const float* hr = h + i * QW;
  const int xi = X[m * LL + i];

  float lg[QW];
  float mx = -3.4e38f;
  float gv = 0.f;
#pragma unroll
  for (int a = 0; a < QW; ++a) {
    lg[a] = pr[a] + hr[a];
    mx = fmaxf(mx, lg[a]);
    gv = (a == xi) ? lg[a] : gv;             // avoid dynamic register indexing
  }
  float se = 0.f;
#pragma unroll
  for (int a = 0; a < QW; ++a) se += expf(lg[a] - mx);
  const float lse = logf(se) + mx;

  red[tid] = -W[m] * (gv - lse);
  __syncthreads();
  for (int off = 128; off > 0; off >>= 1) {
    if (tid < off) red[tid] += red[tid + off];
    __syncthreads();
  }
  if (tid == 0) nllPart[blockIdx.x] = red[0];
}

// ---------------------------------------------------------------------------
// K4: final deterministic reduction: loss = nll + 1e-4*regJ + 1e-6*regH
// ---------------------------------------------------------------------------
__global__ __launch_bounds__(256) void ardca_final(const float* __restrict__ nllPart, int nN,
                                                   const float* __restrict__ partJ, int nJ,
                                                   const float* __restrict__ h, int nH,
                                                   float* __restrict__ out) {
  __shared__ float red[256];
  const int tid = threadIdx.x;
  float sums[3] = {0.f, 0.f, 0.f};
  for (int t = tid; t < nN; t += 256) sums[0] += nllPart[t];
  for (int t = tid; t < nJ; t += 256) sums[1] += partJ[t];
  for (int t = tid; t < nH; t += 256) { const float v = h[t]; sums[2] += v * v; }
  float tot[3];
  for (int q = 0; q < 3; ++q) {
    __syncthreads();
    red[tid] = sums[q];
    __syncthreads();
    for (int off = 128; off > 0; off >>= 1) {
      if (tid < off) red[tid] += red[tid + off];
      __syncthreads();
    }
    tot[q] = red[0];
  }
  if (tid == 0) out[0] = tot[0] + 1e-4f * tot[1] + 1e-6f * tot[2];
}

// ---------------------------------------------------------------------------
extern "C" void kernel_launch(void* const* d_in, const int* in_sizes, int n_in,
                              void* d_out, int out_size, void* d_ws, size_t ws_size,
                              hipStream_t stream) {
  (void)in_sizes; (void)n_in; (void)out_size; (void)ws_size;
  const int*   X = (const int*)d_in[0];     // (M, L) int32
  const float* W = (const float*)d_in[1];   // (M,)
  const float* h = (const float*)d_in[2];   // (L, Q)
  const float* J = (const float*)d_in[3];   // (L, L, Q, Q)
  float* out = (float*)d_out;

  char* ws = (char*)d_ws;
  const size_t JH_BYTES   = (size_t)NK * NK * 2;            // 57.8 MB fp16 permuted J
  const size_t PAIR_BYTES = (size_t)MC * NK * sizeof(float);// 22 MB pair chunk
  _Float16* Jh    = (_Float16*)ws;
  float*    pairc = (float*)(ws + JH_BYTES);
  float*    partJ = (float*)(ws + JH_BYTES + PAIR_BYTES);   // [PREP_BLOCKS]
  float*    nllP  = partJ + PREP_BLOCKS;                    // [M_TOT] (MC slots per chunk)

  ardca_prep<<<PREP_BLOCKS, 256, 0, stream>>>(J, Jh, partJ);

  const int nChunks = M_TOT / MC;
  for (int ch = 0; ch < nChunks; ++ch) {
    const int m0 = ch * MC;
    dim3 g(NK / 128, MC / 128);
    ardca_gemm<<<g, 256, 0, stream>>>(X, Jh, pairc, m0);
    ardca_epi<<<MC, 256, 0, stream>>>(pairc, h, X, W, nllP + ch * MC, m0);
  }
  ardca_final<<<1, 256, 0, stream>>>(nllP, M_TOT, partJ, PREP_BLOCKS, h, NK, out);
}
